// SparseFeatureAttention_50096498541210
// MI455X (gfx1250) — matrix operand order repaired and checked by reading, not compiled, not run on hardware
//
#include <hip/hip_runtime.h>
#include <hip/hip_bf16.h>

#define DEV __device__ __forceinline__

typedef __bf16 bf16_t;
typedef __attribute__((ext_vector_type(16))) __bf16 v16bf;
typedef __attribute__((ext_vector_type(8)))  __bf16 v8bf;
typedef __attribute__((ext_vector_type(8)))  float   v8f;
typedef __attribute__((ext_vector_type(4)))  float   v4f;
typedef __attribute__((ext_vector_type(4)))  unsigned int v4u;
typedef __attribute__((ext_vector_type(4)))  int     v4i;
typedef __attribute__((ext_vector_type(8)))  int     v8i;

union ABf { v16bf v; v8bf h[2]; };

constexpr int   Bn    = 256;
constexpr int   Fn    = 256;
constexpr int   Dn    = 512;
constexpr int   BF    = Bn * Fn;
constexpr int   KK    = 128;                       // int(F * 0.5)
constexpr float SCALE = 0.04419417382415922f;      // 1/sqrt(512)
constexpr int   XP    = 520;                       // padded LDS row stride (bf16): 1040B, conflict-free
constexpr int   AP    = 264;                       // padded attn row stride (bf16): 528B

DEV v8f zero8() {
  v8f z;
  #pragma unroll
  for (int i = 0; i < 8; ++i) z[i] = 0.f;
  return z;
}

DEV v8f wmma_bf16(v16bf a, v16bf b, v8f c) {
  return __builtin_amdgcn_wmma_f32_16x16x32_bf16(
      false, a, false, b, (short)0, c, false, false);
}

DEV void wait_ds0() {
#if __has_builtin(__builtin_amdgcn_s_wait_dscnt)
  __builtin_amdgcn_s_wait_dscnt(0);
#else
  asm volatile("s_wait_dscnt 0" ::: "memory");
#endif
}

// ---------------------------------------------------------------------------
// Kernel 0: one-time fp32 -> bf16 conversion of Wq|Wk|Wv (3 * 512*512)
// ---------------------------------------------------------------------------
__global__ __launch_bounds__(256) void cvt_w_kernel(
    const float* __restrict__ Wq, const float* __restrict__ Wk,
    const float* __restrict__ Wv, bf16_t* __restrict__ wbf)
{
  const size_t n    = (size_t)Dn * Dn;
  const size_t flat = ((size_t)blockIdx.x * 256 + threadIdx.x) * 8;
  const float* s;
  size_t off;
  if (flat < n)            { s = Wq; off = flat; }
  else if (flat < 2 * n)   { s = Wk; off = flat - n; }
  else                     { s = Wv; off = flat - 2 * n; }
  v4f a = *(const v4f*)(s + off);
  v4f b = *(const v4f*)(s + off + 4);
  v8bf o;
  #pragma unroll
  for (int i = 0; i < 4; ++i) { o[i] = (bf16_t)a[i]; o[4 + i] = (bf16_t)b[i]; }
  *(v8bf*)(wbf + flat) = o;
}

// ---------------------------------------------------------------------------
// Kernel 1: q = x Wq^T + bq ; k = x Wk^T + bk ; vT = (x Wv^T + bv)^T
// x row-tile staged once in LDS (bf16, padded rows); weights pre-converted.
// ---------------------------------------------------------------------------
__global__ __launch_bounds__(256) void qkv_kernel(
    const float* __restrict__ x, const bf16_t* __restrict__ wbf,
    const float* __restrict__ bq, const float* __restrict__ bk,
    const float* __restrict__ bv,
    bf16_t* __restrict__ qo, bf16_t* __restrict__ ko, bf16_t* __restrict__ vTo)
{
  __shared__ bf16_t xs[16 * XP];       // staged x tile, bf16, padded rows
  __shared__ bf16_t tbuf[8][16 * 24];  // per-wave 16x16 repack scratch

  const int tid  = threadIdx.x;
  const int wave = tid >> 5;
  const int lane = tid & 31;
  const int ln   = lane & 15;
  const int hi   = lane >> 4;
  const int m0   = blockIdx.x * 16;

  // ---- cooperative stage: 16 rows x 512 f32 -> bf16 LDS (each thread: 32 elems)
  {
    const int r = tid >> 4, c0 = (tid & 15) * 32;
    const float* src = x + (size_t)(m0 + r) * Dn + c0;
    #pragma unroll
    for (int j = 0; j < 4; ++j) {
      v4f f0 = *(const v4f*)(src + j * 8);
      v4f f1 = *(const v4f*)(src + j * 8 + 4);
      v8bf o;
      #pragma unroll
      for (int i = 0; i < 4; ++i) { o[i] = (bf16_t)f0[i]; o[4 + i] = (bf16_t)f1[i]; }
      *(v8bf*)(xs + r * XP + c0 + j * 8) = o;
    }
  }
  __syncthreads();

  v8f acc[3][4];
  #pragma unroll
  for (int i = 0; i < 3; ++i)
    #pragma unroll
    for (int j = 0; j < 4; ++j) acc[i][j] = zero8();

  for (int kit = 0; kit < 16; ++kit) {
    const int k0 = kit * 32;
    ABf A;                                     // A frag from LDS (conflict-free)
    A.h[0] = *(const v8bf*)(xs + ln * XP + k0 + hi * 8);
    A.h[1] = *(const v8bf*)(xs + ln * XP + k0 + 16 + hi * 8);
    __builtin_prefetch(wbf + (size_t)(wave * 64 + ln) * Dn + k0 + 32, 0, 0);
    #pragma unroll
    for (int mat = 0; mat < 3; ++mat) {
      #pragma unroll
      for (int t = 0; t < 4; ++t) {
        const int ncol = wave * 64 + t * 16 + ln;
        const bf16_t* wrow =
            wbf + (size_t)mat * Dn * Dn + (size_t)ncol * Dn + k0 + hi * 8;
        ABf Bv;
        Bv.h[0] = *(const v8bf*)(wrow);
        Bv.h[1] = *(const v8bf*)(wrow + 16);
        acc[mat][t] = wmma_bf16(A.v, Bv.v, acc[mat][t]);
      }
    }
  }

  bf16_t* tb = &tbuf[wave][0];

  // ---- q, k: LDS repack [m][n] -> row-contiguous b128 global stores
  const float* biases[2] = {bq, bk};
  bf16_t*      outs[2]   = {qo, ko};
  #pragma unroll
  for (int mat = 0; mat < 2; ++mat) {
    #pragma unroll 1
    for (int t = 0; t < 4; ++t) {
      const int   nb   = wave * 64 + t * 16;
      const float bias = biases[mat][nb + ln];
      #pragma unroll
      for (int r = 0; r < 8; ++r)                 // elem (m=r+hi*8, n=ln)
        tb[(r + hi * 8) * 24 + ln] = (bf16_t)(acc[mat][t][r] + bias);
      wait_ds0();
      v8bf row = *(const v8bf*)(tb + ln * 24 + hi * 8);   // row m=ln, half hi
      *(v8bf*)(outs[mat] + (size_t)(m0 + ln) * Dn + nb + hi * 8) = row;
      wait_ds0();
    }
  }

  // ---- v: transpose tiles via LDS, store vT[b][d][f] (b128 stores)
  const int bb = m0 / Fn;
  const int f0 = m0 % Fn;
  #pragma unroll 1
  for (int t = 0; t < 4; ++t) {
    const int   nb   = wave * 64 + t * 16;
    const float bias = bv[nb + ln];
    #pragma unroll
    for (int r = 0; r < 8; ++r)                   // elem (m, n=ln) -> tb[n*24+m]
      tb[ln * 24 + r + hi * 8] = (bf16_t)(acc[2][t][r] + bias);
    wait_ds0();
    v8bf rowv = *(const v8bf*)(tb + ln * 24 + hi * 8);    // d-row = ln
    *(v8bf*)(vTo + ((size_t)bb * Dn + nb + ln) * Fn + f0 + hi * 8) = rowv;
    wait_ds0();
  }
}

// ---------------------------------------------------------------------------
// q-tile staging into LDS: TDM tensor_load_to_lds with LDS padding when
// available (pads 16B per 1024B row -> 1040B rows == XP), else manual copy.
// ---------------------------------------------------------------------------
DEV void stage_q_tile(const bf16_t* qb, bf16_t* qs, int tid, int wave) {
#if __has_builtin(__builtin_amdgcn_tensor_load_to_lds) && __has_builtin(__builtin_amdgcn_s_wait_tensorcnt)
  if (wave == 0) {
    union { unsigned int w[4]; v4u v; } g0;
    union { unsigned long long q[4]; v8i v; } g1;
    const unsigned long long ga = (unsigned long long)qb;
    const unsigned int       la = (unsigned int)(unsigned long long)qs; // LDS byte offset
    g0.w[0] = 1u;                                           // count=1
    g0.w[1] = la;                                           // lds_addr
    g0.w[2] = (unsigned int)ga;                             // global_addr[31:0]
    g0.w[3] = (unsigned int)((ga >> 32) & 0x01FFFFFFull) | 0x80000000u; // addr[56:32] | type=2
    // data_size=2B(1), pad_enable, pad_interval=256DW(7), pad_amount=4DW(3),
    // tensor_dim0=512, tensor_dim1=16, tile_dim0=512, tile_dim1=16, stride0=512
    g1.q[0] = (1ull << 16) | (1ull << 20) | (7ull << 22) | (3ull << 25) | (512ull << 48);
    g1.q[1] = (16ull << 16) | (512ull << 48);
    g1.q[2] = 16ull | (512ull << 32);
    g1.q[3] = 0ull;
    v4i z4; z4[0] = z4[1] = z4[2] = z4[3] = 0;
#if defined(__clang_major__) && (__clang_major__ >= 23)
    v8i z8;
    #pragma unroll
    for (int i = 0; i < 8; ++i) z8[i] = 0;
    __builtin_amdgcn_tensor_load_to_lds(g0.v, g1.v, z4, z4, z8, 0);
#else
    __builtin_amdgcn_tensor_load_to_lds(g0.v, g1.v, z4, z4, 0);
#endif
    __builtin_amdgcn_s_wait_tensorcnt(0);
  }
#else
  const int r = tid >> 4, c0 = (tid & 15) * 32;
  #pragma unroll
  for (int j = 0; j < 4; ++j)
    *(v8bf*)(qs + r * XP + c0 + j * 8) =
        *(const v8bf*)(qb + (size_t)r * Dn + c0 + j * 8);
  (void)wave;
#endif
}

// ---------------------------------------------------------------------------
// Kernel 2: fused scores -> top-k threshold -> softmax -> attn @ v
// ---------------------------------------------------------------------------
__global__ __launch_bounds__(256) void attn_kernel(
    const bf16_t* __restrict__ q, const bf16_t* __restrict__ kx,
    const bf16_t* __restrict__ vT, const float* __restrict__ prior,
    float* __restrict__ out)
{
  __shared__ bf16_t qs[16 * XP];      // staged q tile (padded rows)
  __shared__ float  sc[16][256];      // score rows (f32)
  __shared__ bf16_t at[16 * AP];      // normalized attention (padded rows)
  __shared__ float  obuf[8][16 * 20]; // per-wave output repack scratch

  const int tid  = threadIdx.x;
  const int wave = tid >> 5;
  const int lane = tid & 31;
  const int ln   = lane & 15;
  const int hi   = lane >> 4;
  const int bb   = blockIdx.x >> 4;
  const int q0   = (blockIdx.x & 15) * 16;

  const bf16_t* qb = q  + ((size_t)bb * Fn + q0) * Dn;
  const bf16_t* kb = kx + (size_t)bb * Fn * Dn;

  stage_q_tile(qb, qs, tid, wave);
  __syncthreads();

  // ---- scores = q k^T : wave handles key tiles 2*wave, 2*wave+1 (K = 512)
  v8f sacc[2] = {zero8(), zero8()};
  for (int d0 = 0; d0 < Dn; d0 += 32) {
    ABf A;
    A.h[0] = *(const v8bf*)(qs + ln * XP + d0 + hi * 8);
    A.h[1] = *(const v8bf*)(qs + ln * XP + d0 + 16 + hi * 8);
    #pragma unroll
    for (int tl = 0; tl < 2; ++tl) {
      const int key = (wave * 2 + tl) * 16 + ln;
      const bf16_t* krow = kb + (size_t)key * Dn + d0 + hi * 8;
      ABf Bv;
      Bv.h[0] = *(const v8bf*)(krow);
      Bv.h[1] = *(const v8bf*)(krow + 16);
      sacc[tl] = wmma_bf16(A.v, Bv.v, sacc[tl]);
    }
  }
  #pragma unroll
  for (int tl = 0; tl < 2; ++tl) {
    const int kc = (wave * 2 + tl) * 16 + ln;
    #pragma unroll
    for (int r = 0; r < 8; ++r) {
      const int m = r + hi * 8;
      sc[m][kc] = sacc[tl][r] * SCALE + prior[(size_t)(q0 + m) * Fn + kc];
    }
  }
  __syncthreads();

  // ---- per-row: 128th-largest threshold via wave bisection, then softmax
  #pragma unroll 1
  for (int rr = 0; rr < 2; ++rr) {
    const int m = wave * 2 + rr;
    float xv[8];
    v4f x0 = *(const v4f*)(&sc[m][lane * 8]);
    v4f x1 = *(const v4f*)(&sc[m][lane * 8 + 4]);
    #pragma unroll
    for (int i = 0; i < 4; ++i) { xv[i] = x0[i]; xv[4 + i] = x1[i]; }

    float mx = xv[0], mn = xv[0];
    #pragma unroll
    for (int i = 1; i < 8; ++i) { mx = fmaxf(mx, xv[i]); mn = fminf(mn, xv[i]); }
    #pragma unroll
    for (int off = 16; off > 0; off >>= 1) {
      mx = fmaxf(mx, __shfl_xor(mx, off, 32));
      mn = fminf(mn, __shfl_xor(mn, off, 32));
    }

    float lo = mn, hv = mx;
    for (int it = 0; it < 26; ++it) {
      const float mid = 0.5f * (lo + hv);
      int c = 0;
      #pragma unroll
      for (int i = 0; i < 8; ++i) c += (xv[i] >= mid) ? 1 : 0;
      #pragma unroll
      for (int off = 16; off > 0; off >>= 1) c += __shfl_xor(c, off, 32);
      if (c >= KK) lo = mid; else hv = mid;
    }
    const float thr = lo;

    float e[8], s = 0.f;
    #pragma unroll
    for (int i = 0; i < 8; ++i) {
      const float t = (xv[i] >= thr) ? __expf(xv[i] - mx) : 0.f;
      e[i] = t; s += t;
    }
    #pragma unroll
    for (int off = 16; off > 0; off >>= 1) s += __shfl_xor(s, off, 32);
    const float inv = 1.0f / s;
    v8bf eb;
    #pragma unroll
    for (int i = 0; i < 8; ++i) eb[i] = (bf16_t)(e[i] * inv);
    *(v8bf*)(&at[m * AP + lane * 8]) = eb;
  }
  __syncthreads();

  // ---- out = attn @ v (vT gives key-contiguous B fragments)
  v8f oacc[4] = {zero8(), zero8(), zero8(), zero8()};
  for (int kk0 = 0; kk0 < Fn; kk0 += 32) {
    ABf A;
    A.h[0] = *(const v8bf*)(&at[ln * AP + kk0 + hi * 8]);
    A.h[1] = *(const v8bf*)(&at[ln * AP + kk0 + 16 + hi * 8]);
    __builtin_prefetch(vT + ((size_t)bb * Dn + wave * 64 + ln) * Fn + kk0 + 32, 0, 0);
    #pragma unroll
    for (int t = 0; t < 4; ++t) {
      const int d = wave * 64 + t * 16 + ln;
      const bf16_t* vrow = vT + ((size_t)bb * Dn + d) * Fn + kk0 + hi * 8;
      ABf Bv;
      Bv.h[0] = *(const v8bf*)(vrow);
      Bv.h[1] = *(const v8bf*)(vrow + 16);
      oacc[t] = wmma_bf16(A.v, Bv.v, oacc[t]);
    }
  }

  // ---- output: per-wave LDS repack -> b128 coalesced f32 stores
  float* ob = &obuf[wave][0];
  #pragma unroll 1
  for (int t = 0; t < 4; ++t) {
    const int nb = wave * 64 + t * 16;
    #pragma unroll
    for (int r = 0; r < 8; ++r)                   // elem (m=r+hi*8, n=ln)
      ob[(r + hi * 8) * 20 + ln] = oacc[t][r];
    wait_ds0();
    v4f r0 = *(const v4f*)(ob + ln * 20 + hi * 8);
    v4f r1 = *(const v4f*)(ob + ln * 20 + hi * 8 + 4);
    float* dst = out + ((size_t)bb * Fn + q0 + ln) * Dn + nb + hi * 8;
    *(v4f*)(dst)     = r0;
    *(v4f*)(dst + 4) = r1;
    wait_ds0();
  }
}

// ---------------------------------------------------------------------------
extern "C" void kernel_launch(void* const* d_in, const int* in_sizes, int n_in,
                              void* d_out, int out_size, void* d_ws, size_t ws_size,
                              hipStream_t stream) {
  (void)in_sizes; (void)n_in; (void)out_size; (void)ws_size;

  const float* x     = (const float*)d_in[0];
  const float* Wq    = (const float*)d_in[1];
  const float* bq    = (const float*)d_in[2];
  const float* Wk    = (const float*)d_in[3];
  const float* bk    = (const float*)d_in[4];
  const float* Wv    = (const float*)d_in[5];
  const float* bv    = (const float*)d_in[6];
  const float* prior = (const float*)d_in[7];
  float*       out   = (float*)d_out;

  // ws: q | k | vT (each B*F*D bf16 = 64MB) | Wbf16 (3*D*D bf16 = 1.5MB)
  const size_t elems = (size_t)BF * Dn;
  bf16_t* qws  = (bf16_t*)d_ws;
  bf16_t* kws  = qws + elems;
  bf16_t* vTws = kws + elems;
  bf16_t* wbf  = vTws + elems;

  cvt_w_kernel<<<(3 * Dn * Dn) / (256 * 8), 256, 0, stream>>>(Wq, Wk, Wv, wbf);
  qkv_kernel<<<BF / 16, 256, 0, stream>>>(x, wbf, bq, bk, bv, qws, kws, vTws);
  attn_kernel<<<Bn * (Fn / 16), 256, 0, stream>>>(qws, kws, vTws, prior, out);
}
